// KPCR_S_120259084572
// MI455X (gfx1250) — matrix operand order repaired and checked by reading, not compiled, not run on hardware
//
#include <hip/hip_runtime.h>
#include <hip/hip_bf16.h>

typedef __attribute__((ext_vector_type(16))) __bf16 v16bf;
typedef __attribute__((ext_vector_type(8)))  __bf16 v8bf;
typedef __attribute__((ext_vector_type(8)))  float  v8f;

#define B_      512
#define NENT    100000
#define D_      200
#define KPAD    224
#define C_      32
#define HW      324          // 18*18
#define FCIN    10368        // 32*324
#define EPS_    1e-5f

// ---- workspace layout (bytes) ----
#define HDR_BYTES   4096
#define OFF_XCAT    4096                         // 512*400 f32
#define OFF_CONV    (OFF_XCAT + 819200)          // 512*10368 f32
#define OFF_FEAT    (OFF_CONV + 21233664)        // 512*10368 bf16
#define OFF_OUT1    (OFF_FEAT + 10616832)        // 512*200 f32
#define OFF_X2F     (OFF_OUT1 + 409600)          // 512*200 f32
#define OFF_X2BF    (OFF_X2F + 409600)           // 512*224 bf16

// header float indices (base = ws + 64)
#define F_BN0SUM   0
#define F_SSQ_H    1
#define F_SSQ_R    2
#define F_BN0A     3
#define F_BN0B     4
#define F_FCSSQ    5
#define F_CWSSQ    6
#define F_BN1SUM   8     // 32
#define F_BN1SSQ   40    // 32
#define F_BN1SC    72    // 32
#define F_BN1SH    104   // 32
#define F_BN2SC    136   // 200
#define F_BN2SH    336   // 200

__device__ __forceinline__ float wave_sum(float v) {
#pragma unroll
  for (int o = 16; o > 0; o >>= 1) v += __shfl_xor(v, o, 32);
  return v;
}

// ---- WMMA fragment loaders ------------------------------------------------
// A (16x32 bf16, MxK): lanes 0-15 row M=lane hold K=k0+0..15;
// lanes 16-31 row M=lane-16 hold K=k0+16..31.
__device__ __forceinline__ v16bf load_a_frag(const __bf16* base, int stride,
                                             int m0, int k0, int lane) {
  int m  = m0 + (lane & 15);
  int kb = k0 + ((lane >> 4) << 4);
  const __bf16* p = base + m * stride + kb;
  union { v16bf v; v8bf h[2]; } u;
  u.h[0] = *(const v8bf*)(p);
  u.h[1] = *(const v8bf*)(p + 8);
  return u.v;
}

// B (32x16 bf16, KxN) from a row-major f32 [N][K] matrix, converting in
// registers: lane holds column nrow (pre-clamped by caller; garbage columns
// are simply never stored), K = k0 + (lane>>4)*16 + 0..15.
__device__ __forceinline__ v16bf load_b_frag_cvt(const float* base, int stride,
                                                 int nrow, int k0, int lane,
                                                 int kmax) {
  int kb = k0 + ((lane >> 4) << 4);
  const float* p = base + (long long)nrow * stride + kb;
  v16bf out;
#pragma unroll
  for (int j = 0; j < 16; ++j) {
    float v = (kb + j < kmax) ? p[j] : 0.0f;
    out[j] = (__bf16)v;
  }
  return out;
}

// ---- kernels ---------------------------------------------------------------

__global__ void k_init(int* hdr) {
  int i = threadIdx.x;
  if (i < 1024) hdr[i] = 0;
}

// gather embeddings -> xcat, accumulate BN0 sum / sumsq (h and r parts)
__global__ void k_gather_bn0(const float* __restrict__ ent,
                             const float* __restrict__ rel,
                             const int* __restrict__ h, const int* __restrict__ r,
                             float* __restrict__ xcat, float* f) {
  int idx = blockIdx.x * blockDim.x + threadIdx.x;   // 512*400 exact
  int b = idx / 400, k = idx % 400;
  float v = (k < 200) ? ent[(long long)h[b] * 200 + k]
                      : rel[(long long)r[b] * 200 + (k - 200)];
  xcat[idx] = v;
  float s = v, qh = (k < 200) ? v * v : 0.0f, qr = (k < 200) ? 0.0f : v * v;
  s = wave_sum(s); qh = wave_sum(qh); qr = wave_sum(qr);
  __shared__ float s1[8], s2[8], s3[8];
  int wid = threadIdx.x >> 5, lane = threadIdx.x & 31;
  if (lane == 0) { s1[wid] = s; s2[wid] = qh; s3[wid] = qr; }
  __syncthreads();
  if (threadIdx.x == 0) {
    float t1 = 0, t2 = 0, t3 = 0;
    for (int i = 0; i < 8; ++i) { t1 += s1[i]; t2 += s2[i]; t3 += s3[i]; }
    atomicAdd(&f[F_BN0SUM], t1);
    atomicAdd(&f[F_SSQ_H], t2);
    atomicAdd(&f[F_SSQ_R], t3);
  }
}

__global__ void k_bn0_final(float* f, const float* g0, const float* b0) {
  float n = (float)(B_ * 400);
  float m = f[F_BN0SUM] / n;
  float var = (f[F_SSQ_H] + f[F_SSQ_R]) / n - m * m;
  float sc = g0[0] * rsqrtf(var + EPS_);
  f[F_BN0A] = sc;
  f[F_BN0B] = b0[0] - m * sc;
}

// 3x3 valid conv on BN0(xcat) with BN0 folded: conv(A*x+B) = A*conv(x)+B*sum(w)
// one block per (b,c) plane; accumulates per-channel BN1 stats
__global__ void k_conv(const float* __restrict__ xcat,
                       const float* __restrict__ cw, const float* __restrict__ cb,
                       const float* __restrict__ f, float* __restrict__ conv_raw) {
  int bc = blockIdx.x;
  int b = bc >> 5, c = bc & 31;
  float w[9], wsum = 0.0f;
#pragma unroll
  for (int i = 0; i < 9; ++i) { w[i] = cw[c * 9 + i]; wsum += w[i]; }
  float A = f[F_BN0A], Bc = f[F_BN0B] * wsum + cb[c];
  const float* xrow = xcat + b * 400;
  float ls = 0.0f, lq = 0.0f;
  for (int idx = threadIdx.x; idx < HW; idx += blockDim.x) {
    int y = idx / 18, x = idx % 18;
    float pre = 0.0f;
#pragma unroll
    for (int ky = 0; ky < 3; ++ky) {
      const float* rp = xrow + (y + ky) * 20 + x;
      pre += w[ky * 3 + 0] * rp[0] + w[ky * 3 + 1] * rp[1] + w[ky * 3 + 2] * rp[2];
    }
    float val = A * pre + Bc;
    conv_raw[(long long)b * FCIN + c * HW + idx] = val;
    ls += val; lq += val * val;
  }
  ls = wave_sum(ls); lq = wave_sum(lq);
  __shared__ float s1[4], s2[4];
  int wid = threadIdx.x >> 5, lane = threadIdx.x & 31;
  if (lane == 0) { s1[wid] = ls; s2[wid] = lq; }
  __syncthreads();
  if (threadIdx.x == 0) {
    float* fm = (float*)f;   // header is writable
    atomicAdd(&fm[F_BN1SUM + c], s1[0] + s1[1] + s1[2] + s1[3]);
    atomicAdd(&fm[F_BN1SSQ + c], s2[0] + s2[1] + s2[2] + s2[3]);
  }
}

// finalize BN1 scale/shift + conv_w L2 sumsq (single block of 512)
__global__ void k_bn1_final(float* f, const float* g1, const float* b1,
                            const float* cw) {
  int t = threadIdx.x;
  if (t < 32) {
    float n = (float)(B_ * HW);
    float m = f[F_BN1SUM + t] / n;
    float var = f[F_BN1SSQ + t] / n - m * m;
    float sc = g1[t] * rsqrtf(var + EPS_);
    f[F_BN1SC + t] = sc;
    f[F_BN1SH + t] = b1[t] - m * sc;
  }
  __shared__ float red[512];
  float sq = 0.0f;
  if (t < 288) { float w = cw[t]; sq = w * w; }
  red[t] = sq;
  __syncthreads();
  for (int s = 256; s > 0; s >>= 1) {
    if (t < s) red[t] += red[t + s];
    __syncthreads();
  }
  if (t == 0) f[F_CWSSQ] = red[0];
}

// fc_w L2 sumsq
__global__ void k_fc_l2(const float* __restrict__ fcw, float* f) {
  float s = 0.0f;
  for (long long i = blockIdx.x * (long long)blockDim.x + threadIdx.x;
       i < (long long)D_ * FCIN; i += (long long)gridDim.x * blockDim.x) {
    float w = fcw[i]; s += w * w;
  }
  s = wave_sum(s);
  __shared__ float sh[8];
  int wid = threadIdx.x >> 5, lane = threadIdx.x & 31;
  if (lane == 0) sh[wid] = s;
  __syncthreads();
  if (threadIdx.x == 0) {
    float t = 0; for (int i = 0; i < 8; ++i) t += sh[i];
    atomicAdd(&f[F_FCSSQ], t);
  }
}

// BN1 + ReLU -> bf16 features
__global__ void k_feat(const float* __restrict__ conv_raw,
                       const float* __restrict__ f, __bf16* __restrict__ feat) {
  long long idx = blockIdx.x * (long long)blockDim.x + threadIdx.x; // exact
  int k = (int)(idx % FCIN);
  int c = k / HW;
  float v = conv_raw[idx];
  v = fmaxf(v * f[F_BN1SC + c] + f[F_BN1SH + c], 0.0f);
  feat[idx] = (__bf16)v;
}

// FC GEMM: out1[m,n] = feat[m,:] . fc_w[n,:] + fc_b[n]   (M=512,N=200,K=10368)
// lanes whose column exceeds 199 clamp to row 199: they compute garbage
// columns that are never stored, keeping the inner loop branch-free.
__global__ void k_fc_gemm(const __bf16* __restrict__ feat,
                          const float* __restrict__ fcw,
                          const float* __restrict__ fcb,
                          float* __restrict__ out1) {
  int gwid = blockIdx.x * (blockDim.x >> 5) + (threadIdx.x >> 5);
  if (gwid >= 32 * 13) return;
  int mt = gwid / 13, nt = gwid % 13;
  int m0 = mt * 16, n0 = nt * 16;
  int lane = threadIdx.x & 31;
  int n = n0 + (lane & 15);
  int nc = (n < D_) ? n : (D_ - 1);
  v8f c = {0.f, 0.f, 0.f, 0.f, 0.f, 0.f, 0.f, 0.f};
  for (int kt = 0; kt < FCIN / 32; ++kt) {
    int k0 = kt * 32;
    v16bf a = load_a_frag(feat, FCIN, m0, k0, lane);
    v16bf bb = load_b_frag_cvt(fcw, FCIN, nc, k0, lane, FCIN);
    c = __builtin_amdgcn_wmma_f32_16x16x32_bf16(false, a, false, bb,
                                                (short)0, c, false, false);
  }
  if (n < D_) {
    float bias = fcb[n];
    int mb = m0 + ((lane >> 4) << 3);
#pragma unroll
    for (int j = 0; j < 8; ++j) out1[(mb + j) * D_ + n] = c[j] + bias;
  }
}

// BN2 stats + scale/shift (one block per feature)
__global__ void k_bn2(const float* __restrict__ out1, const float* g2,
                      const float* b2, float* f) {
  int d = blockIdx.x;
  float s = 0.0f, q = 0.0f;
  for (int row = threadIdx.x; row < B_; row += blockDim.x) {
    float v = out1[row * D_ + d]; s += v; q += v * v;
  }
  s = wave_sum(s); q = wave_sum(q);
  __shared__ float s1[8], s2[8];
  int wid = threadIdx.x >> 5, lane = threadIdx.x & 31;
  if (lane == 0) { s1[wid] = s; s2[wid] = q; }
  __syncthreads();
  if (threadIdx.x == 0) {
    float t1 = 0, t2 = 0;
    for (int i = 0; i < 8; ++i) { t1 += s1[i]; t2 += s2[i]; }
    float m = t1 / (float)B_;
    float var = t2 / (float)B_ - m * m;
    float sc = g2[d] * rsqrtf(var + EPS_);
    f[F_BN2SC + d] = sc;
    f[F_BN2SH + d] = b2[d] - m * sc;
  }
}

// BN2 + leaky-relu -> x2 (f32 for pos-dots, bf16 zero-padded to K=224)
__global__ void k_x2(const float* __restrict__ out1, const float* __restrict__ f,
                     float* __restrict__ x2f, __bf16* __restrict__ x2bf) {
  int idx = blockIdx.x * blockDim.x + threadIdx.x;   // 512*224 exact
  int b = idx / KPAD, k = idx % KPAD;
  if (k < D_) {
    float v = out1[b * D_ + k] * f[F_BN2SC + k] + f[F_BN2SH + k];
    v = (v > 0.0f) ? v : 0.01f * v;
    x2f[b * D_ + k] = v;
    x2bf[idx] = (__bf16)v;
  } else {
    x2bf[idx] = (__bf16)0.0f;
  }
}

// score GEMM fused with softplus/sum epilogue; scores never hit memory.
// each wave owns TWO 16-entity strips (B frags f32->bf16 in regs, held in
// VGPRs across all 32 M-tiles): each A fragment feeds two WMMAs, halving
// A traffic and doubling matrix work per load.
__global__ void k_score(const __bf16* __restrict__ x2bf,
                        const float* __restrict__ ent,
                        const float* __restrict__ bvec, double* acc) {
  int gwid = blockIdx.x * (blockDim.x >> 5) + (threadIdx.x >> 5);
  int e0 = gwid * 32;
  if (e0 >= NENT) return;
  int lane = threadIdx.x & 31;
  int ncol = lane & 15;
  v16bf bf0[7], bf1[7];
#pragma unroll
  for (int kt = 0; kt < 7; ++kt) {
    bf0[kt] = load_b_frag_cvt(ent, D_, e0 + ncol,      kt * 32, lane, D_);
    bf1[kt] = load_b_frag_cvt(ent, D_, e0 + 16 + ncol, kt * 32, lane, D_);
  }
  float bias0 = bvec[e0 + ncol];
  float bias1 = bvec[e0 + 16 + ncol];
  float sp = 0.0f, ss = 0.0f;
  for (int mt = 0; mt < 32; ++mt) {
    int m0 = mt * 16;
    v8f c0 = {0.f, 0.f, 0.f, 0.f, 0.f, 0.f, 0.f, 0.f};
    v8f c1 = {0.f, 0.f, 0.f, 0.f, 0.f, 0.f, 0.f, 0.f};
#pragma unroll
    for (int kt = 0; kt < 7; ++kt) {
      v16bf a = load_a_frag(x2bf, KPAD, m0, kt * 32, lane);
      c0 = __builtin_amdgcn_wmma_f32_16x16x32_bf16(false, a, false, bf0[kt],
                                                   (short)0, c0, false, false);
      c1 = __builtin_amdgcn_wmma_f32_16x16x32_bf16(false, a, false, bf1[kt],
                                                   (short)0, c1, false, false);
    }
#pragma unroll
    for (int j = 0; j < 8; ++j) {
      float s0 = c0[j] + bias0;
      float s1 = c1[j] + bias1;
      ss += s0 + s1;
      sp += fmaxf(s0, 0.0f) + __logf(1.0f + __expf(-fabsf(s0)));
      sp += fmaxf(s1, 0.0f) + __logf(1.0f + __expf(-fabsf(s1)));
    }
  }
  sp = wave_sum(sp);
  ss = wave_sum(ss);
  if (lane == 0) {
    atomicAdd(&acc[0], (double)sp);
    atomicAdd(&acc[1], (double)ss);
  }
}

// positive-target scores in f32: scores[i, pos_t[i]]
__global__ void k_pos(const float* __restrict__ x2f, const float* __restrict__ ent,
                      const float* __restrict__ bvec, const int* __restrict__ pos,
                      double* acc) {
  int i = blockIdx.x;
  int t = pos[i];
  float s = 0.0f;
  for (int d = threadIdx.x; d < D_; d += 64)
    s += x2f[i * D_ + d] * ent[(long long)t * D_ + d];
  s = wave_sum(s);
  __shared__ float sh[2];
  if ((threadIdx.x & 31) == 0) sh[threadIdx.x >> 5] = s;
  __syncthreads();
  if (threadIdx.x == 0)
    atomicAdd(&acc[2], (double)(sh[0] + sh[1] + bvec[t]));
}

__global__ void k_final(const double* acc, const float* f, float* out) {
  double sp = acc[0], ssum = acc[1], pos = acc[2];
  double xy = ssum / (double)NENT + (1.0 - 0.1) * pos;
  double kg = (sp - xy) / ((double)B_ * (double)NENT);
  double l2 = (double)f[F_SSQ_H] / (2.0 * B_ * D_) +
              (double)f[F_SSQ_R] / (2.0 * B_ * D_) +
              (double)f[F_CWSSQ] / (2.0 * 288.0) +
              (double)f[F_FCSSQ] / (2.0 * 200.0);
  out[0] = (float)(kg + 1e-5 * l2);
}

extern "C" void kernel_launch(void* const* d_in, const int* in_sizes, int n_in,
                              void* d_out, int out_size, void* d_ws, size_t ws_size,
                              hipStream_t stream) {
  (void)in_sizes; (void)n_in; (void)out_size; (void)ws_size;
  const float* ent  = (const float*)d_in[0];
  const float* rel  = (const float*)d_in[1];
  const float* cw   = (const float*)d_in[2];
  const float* cb   = (const float*)d_in[3];
  const float* fcw  = (const float*)d_in[4];
  const float* fcb  = (const float*)d_in[5];
  const float* bvec = (const float*)d_in[6];
  const float* g0 = (const float*)d_in[7],  *b0 = (const float*)d_in[8];
  const float* g1 = (const float*)d_in[9],  *b1 = (const float*)d_in[10];
  const float* g2 = (const float*)d_in[11], *b2 = (const float*)d_in[12];
  const int* h   = (const int*)d_in[13];
  const int* r   = (const int*)d_in[14];
  const int* pos = (const int*)d_in[15];

  char* ws = (char*)d_ws;
  double* acc = (double*)ws;
  float* f = (float*)(ws + 64);
  float*  xcat = (float*)(ws + OFF_XCAT);
  float*  conv_raw = (float*)(ws + OFF_CONV);
  __bf16* feat = (__bf16*)(ws + OFF_FEAT);
  float*  out1 = (float*)(ws + OFF_OUT1);
  float*  x2f  = (float*)(ws + OFF_X2F);
  __bf16* x2bf = (__bf16*)(ws + OFF_X2BF);
  float*  outp = (float*)d_out;

  hipLaunchKernelGGL(k_init, dim3(1), dim3(1024), 0, stream, (int*)ws);
  hipLaunchKernelGGL(k_gather_bn0, dim3(800), dim3(256), 0, stream,
                     ent, rel, h, r, xcat, f);
  hipLaunchKernelGGL(k_bn0_final, dim3(1), dim3(1), 0, stream, f, g0, b0);
  hipLaunchKernelGGL(k_fc_l2, dim3(480), dim3(256), 0, stream, fcw, f);
  hipLaunchKernelGGL(k_conv, dim3(B_ * C_), dim3(128), 0, stream,
                     xcat, cw, cb, f, conv_raw);
  hipLaunchKernelGGL(k_bn1_final, dim3(1), dim3(512), 0, stream, f, g1, b1, cw);
  hipLaunchKernelGGL(k_feat, dim3((B_ * FCIN) / 256), dim3(256), 0, stream,
                     conv_raw, f, feat);
  hipLaunchKernelGGL(k_fc_gemm, dim3(52), dim3(256), 0, stream,
                     feat, fcw, fcb, out1);
  hipLaunchKernelGGL(k_bn2, dim3(D_), dim3(256), 0, stream, out1, g2, b2, f);
  hipLaunchKernelGGL(k_x2, dim3((B_ * KPAD) / 256), dim3(256), 0, stream,
                     out1, f, x2f, x2bf);
  // 100000/32 = 3125 waves, 8 waves per block
  hipLaunchKernelGGL(k_score, dim3(391), dim3(256), 0, stream,
                     x2bf, ent, bvec, acc);
  hipLaunchKernelGGL(k_pos, dim3(B_), dim3(64), 0, stream,
                     x2f, ent, bvec, pos, acc);
  hipLaunchKernelGGL(k_final, dim3(1), dim3(1), 0, stream, acc, f, outp);
}
